// InputAttnEncoder_17231408791856
// MI455X (gfx1250) — matrix operand order repaired and checked by reading, not compiled, not run on hardware
//
#include <hip/hip_runtime.h>
#include <hip/hip_bf16.h>
#include <stdint.h>

// ---------------------------------------------------------------------------
// InputAttnEncoder for MI455X (gfx1250): bf16 WMMA (16x16x32) f32-accumulate,
// TDM (tensor_load_to_lds) staging of GEMM A-panels into LDS.
// B=1024, T=83, F=512, H=512.
// ---------------------------------------------------------------------------

#define B_SZ 1024
#define T_SZ 83
#define F_SZ 512
#define H_SZ 512
#define K1 1536      // F + 2H (concat dim)
#define N1P 96       // T+1=84 padded to 96
#define G4 2048      // 4H
#define KG 1024      // F + H (xi | h)

typedef __bf16 bf16;
typedef bf16  bf16x16 __attribute__((ext_vector_type(16)));
typedef float floatx8 __attribute__((ext_vector_type(8)));
typedef uint32_t u32x4 __attribute__((ext_vector_type(4)));
typedef int      i32x8 __attribute__((ext_vector_type(8)));
typedef int      i32x4 __attribute__((ext_vector_type(4)));

union FragBF { bf16x16 v; uint4 q[2]; };

#define WMMA_BF16(A, Bm, C) \
  __builtin_amdgcn_wmma_f32_16x16x32_bf16(false, (A), false, (Bm), (short)0, (C), false, false)

__device__ __forceinline__ bf16 f2bf(float x) {
  union { float f; uint32_t u; } v; v.f = x;
  uint32_t r = v.u + 0x7FFFu + ((v.u >> 16) & 1u);
  union { uint16_t s; bf16 b; } o; o.s = (uint16_t)(r >> 16);
  return o.b;
}

// A fragment: 16x32 bf16 tile (MxK) from row-major [rows, ld].
// Lane L: m = L&15, half = L>>4. VGPR j<4 -> K = k0+8*half+2j..+1 ; j>=4 -> +16.
__device__ __forceinline__ bf16x16 load_a_frag(const bf16* A, int ld, int row0, int k0, int lane) {
  int m = lane & 15, half = lane >> 4;
  const bf16* p = A + (size_t)(row0 + m) * ld + k0 + 8 * half;
  FragBF f;
  f.q[0] = *(const uint4*)(p);
  f.q[1] = *(const uint4*)(p + 16);
  return f.v;
}

// B fragment: 32x16 bf16 tile (KxN) where B = W^T, W row-major [N, ld].
// Lane L: n = L&15, half = L>>4. VGPR j<4 -> K = k0+8*half+2j..+1 ; j>=4 -> +16.
__device__ __forceinline__ bf16x16 load_b_frag(const bf16* W, int ld, int n0, int k0, int lane) {
  int n = lane & 15, half = lane >> 4;
  const bf16* p = W + (size_t)(n0 + n) * ld + k0 + 8 * half;
  FragBF f;
  f.q[0] = *(const uint4*)(p);
  f.q[1] = *(const uint4*)(p + 16);
  return f.v;
}

// Stage a [rows x cols] bf16 panel (row-major, contiguous, stride==cols) from
// global into LDS using the Tensor Data Mover. Call from ONE wave; follow with
// __syncthreads() in the caller. D# layout per CDNA5 ISA.
// Toolchain uses the 6-arg builtin: (g0, g1, g2, g3, g4, cpol).
__device__ __forceinline__ void tdm_stage_panel(const bf16* gsrc, bf16* lds_dst,
                                                int rows, int cols) {
#if __has_builtin(__builtin_amdgcn_tensor_load_to_lds)
  uint64_t gaddr = (uint64_t)(uintptr_t)gsrc;
  uint32_t ldsa = (uint32_t)(uintptr_t)lds_dst;  // low 32 bits = LDS byte offset
  u32x4 g0; i32x8 g1; i32x4 g2 = {0, 0, 0, 0}, g3 = {0, 0, 0, 0};
  i32x8 g4 = {0, 0, 0, 0, 0, 0, 0, 0};
  g0[0] = 1u;                                     // count=1 (valid), user mode
  g0[1] = ldsa;                                   // lds_addr (bytes)
  g0[2] = (uint32_t)gaddr;                        // global_addr[31:0]
  g0[3] = (uint32_t)(gaddr >> 32) | (2u << 30);   // global_addr[56:32] | type=2
  g1[0] = (int)(1u << 16);                        // wg_mask=0, data_size=1 (2B)
  g1[1] = (int)((cols & 0xFFFF) << 16);           // tensor_dim0[15:0]
  g1[2] = (int)(((uint32_t)cols >> 16) | ((rows & 0xFFFF) << 16)); // dim0 hi | dim1 lo
  g1[3] = (int)((cols & 0xFFFF) << 16);           // dim1 hi (=0) | tile_dim0
  g1[4] = rows;                                   // tile_dim1 | tile_dim2=0
  g1[5] = cols;                                   // tensor_dim0_stride[31:0]
  g1[6] = 0;                                      // stride0 hi | dim1_stride lo
  g1[7] = 0;
  __builtin_amdgcn_tensor_load_to_lds(g0, g1, g2, g3, g4, 0);
  __builtin_amdgcn_s_wait_tensorcnt(0);
#else
  // fallback: cooperative wide-copy by the calling wave
  for (int i = (int)(threadIdx.x & 31); i < rows * cols / 8; i += 32)
    ((uint4*)lds_dst)[i] = ((const uint4*)gsrc)[i];
#endif
}

// ----------------------- one-time weight conversion ------------------------

__global__ void k_cvt_padcol(const float* __restrict__ src, bf16* __restrict__ dst,
                             int rows, int cols_src, int cols_dst) {
  int idx = blockIdx.x * blockDim.x + threadIdx.x;
  if (idx >= rows * cols_dst) return;
  int r = idx / cols_dst, c = idx % cols_dst;
  dst[idx] = (c < cols_src) ? f2bf(src[(size_t)r * cols_src + c]) : f2bf(0.f);
}

__global__ void k_cvt_padrow(const float* __restrict__ src, bf16* __restrict__ dst,
                             int rows_src, int rows_dst, int cols) {
  int idx = blockIdx.x * blockDim.x + threadIdx.x;
  if (idx >= rows_dst * cols) return;
  int r = idx / cols, c = idx % cols;
  dst[idx] = (r < rows_src) ? f2bf(src[(size_t)r * cols + c]) : f2bf(0.f);
}

__global__ void k_cvt_wg(const float* __restrict__ Wih, const float* __restrict__ Whh,
                         bf16* __restrict__ dst) {
  int idx = blockIdx.x * blockDim.x + threadIdx.x;
  if (idx >= G4 * KG) return;
  int n = idx >> 10, k = idx & 1023;
  float v = (k < F_SZ) ? Wih[(size_t)n * F_SZ + k] : Whh[(size_t)n * H_SZ + (k - F_SZ)];
  dst[idx] = f2bf(v);
}

__global__ void k_init(bf16* __restrict__ catA, bf16* __restrict__ xg, float* __restrict__ c_f32) {
  int idx = blockIdx.x * blockDim.x + threadIdx.x;
  if (idx >= B_SZ * 1024) return;
  int b = idx >> 10, j = idx & 1023;
  catA[(size_t)b * K1 + F_SZ + j] = f2bf(0.f);
  if (j < H_SZ) {
    xg[(size_t)b * KG + F_SZ + j] = f2bf(0.f);
    c_f32[(size_t)b * H_SZ + j] = 0.f;
  }
}

__global__ void k_prep(const float* __restrict__ X, bf16* __restrict__ catA, int t) {
  int idx = blockIdx.x * blockDim.x + threadIdx.x;
  if (idx >= B_SZ * F_SZ) return;
  int b = idx >> 9, f = idx & 511;
  catA[(size_t)b * K1 + f] = f2bf(X[((size_t)b * T_SZ + t) * F_SZ + f]);
}

// ------------------------- fused attention block ---------------------------
__global__ __launch_bounds__(256) void k_attn(
    const bf16* __restrict__ catA, const bf16* __restrict__ Wb1, const float* __restrict__ b1,
    const bf16* __restrict__ Wb2, const float* __restrict__ b2,
    const float* __restrict__ X, float* __restrict__ xi_f32, bf16* __restrict__ xg, int t) {
  __shared__ __align__(16) bf16 xi1_s[32][N1P];
  const int lane = threadIdx.x & 31;
  const int w = threadIdx.x >> 5;
  const int row0 = blockIdx.x * 32;
  const int n = lane & 15, half = lane >> 4;

  if (w < 6) {  // phase 1: 6 N-tiles of 16 over 96 padded cols
    const int n0 = w * 16;
    floatx8 acc0 = {}, acc1 = {};
    for (int kt = 0; kt < K1 / 32; ++kt) {
      int k0 = kt * 32;
      bf16x16 a0 = load_a_frag(catA, K1, row0, k0, lane);
      bf16x16 a1 = load_a_frag(catA, K1, row0 + 16, k0, lane);
      bf16x16 fb = load_b_frag(Wb1, K1, n0, k0, lane);
      acc0 = WMMA_BF16(a0, fb, acc0);
      acc1 = WMMA_BF16(a1, fb, acc1);
    }
    int col = n0 + n;
    float bias = (col < T_SZ + 1) ? b1[col] : 0.0f;
#pragma unroll
    for (int j = 0; j < 8; ++j) {
      float v0 = acc0[j] + bias; v0 = v0 > 0.f ? v0 : 0.f;
      float v1 = acc1[j] + bias; v1 = v1 > 0.f ? v1 : 0.f;
      xi1_s[j + 8 * half][col] = f2bf(v0);
      xi1_s[16 + j + 8 * half][col] = f2bf(v1);
    }
  }
  __syncthreads();

  // phase 2: each wave covers 4 N-tiles x 2 M-tiles; K = 96 (3 k-steps)
  floatx8 acc[2][4];
#pragma unroll
  for (int mt = 0; mt < 2; ++mt)
#pragma unroll
    for (int i = 0; i < 4; ++i) acc[mt][i] = (floatx8){};

  for (int kt = 0; kt < 3; ++kt) {
    int k0 = kt * 32;
    bf16x16 a0 = load_a_frag(&xi1_s[0][0], N1P, 0, k0, lane);
    bf16x16 a1 = load_a_frag(&xi1_s[0][0], N1P, 16, k0, lane);
#pragma unroll
    for (int i = 0; i < 4; ++i) {
      bf16x16 fb = load_b_frag(Wb2, N1P, w * 64 + i * 16, k0, lane);
      acc[0][i] = WMMA_BF16(a0, fb, acc[0][i]);
      acc[1][i] = WMMA_BF16(a1, fb, acc[1][i]);
    }
  }
#pragma unroll
  for (int mt = 0; mt < 2; ++mt) {
#pragma unroll
    for (int i = 0; i < 4; ++i) {
      int col = w * 64 + i * 16 + n;
      float bias = b2[col];
#pragma unroll
      for (int j = 0; j < 8; ++j) {
        int row = row0 + mt * 16 + j + 8 * half;
        float xi2 = acc[mt][i][j] + bias;
        float xv = X[((size_t)row * T_SZ + t) * F_SZ + col];
        float xiv = xi2 * xv;
        xi_f32[(size_t)row * F_SZ + col] = xiv;
        xg[(size_t)row * KG + col] = f2bf(xiv);
      }
    }
  }
}

// gates = [xi | h] @ Wg^T + b_ih + b_hh. TDM stages the 32x1024 A panel in LDS.
__global__ __launch_bounds__(256) void k_gates(
    const bf16* __restrict__ xg, const bf16* __restrict__ Wg,
    const float* __restrict__ b_ih, const float* __restrict__ b_hh,
    float* __restrict__ gates) {
  __shared__ __align__(16) bf16 lds_a[32 * KG];  // 64 KB
  const int lane = threadIdx.x & 31, w = threadIdx.x >> 5;
  const int row0 = blockIdx.x * 32;
  const int n_base = blockIdx.y * 512 + w * 64;  // 4 N-tiles per wave
  const int n = lane & 15, half = lane >> 4;

  if (w == 0) tdm_stage_panel(xg + (size_t)row0 * KG, lds_a, 32, KG);
  __syncthreads();

  floatx8 acc[2][4];
#pragma unroll
  for (int mt = 0; mt < 2; ++mt)
#pragma unroll
    for (int i = 0; i < 4; ++i) acc[mt][i] = (floatx8){};

  for (int kt = 0; kt < KG / 32; ++kt) {
    int k0 = kt * 32;
    bf16x16 a0 = load_a_frag(lds_a, KG, 0, k0, lane);
    bf16x16 a1 = load_a_frag(lds_a, KG, 16, k0, lane);
#pragma unroll
    for (int i = 0; i < 4; ++i) {
      bf16x16 fb = load_b_frag(Wg, KG, n_base + i * 16, k0, lane);
      acc[0][i] = WMMA_BF16(a0, fb, acc[0][i]);
      acc[1][i] = WMMA_BF16(a1, fb, acc[1][i]);
    }
  }
#pragma unroll
  for (int i = 0; i < 4; ++i) {
    int col = n_base + i * 16 + n;
    float bias = b_ih[col] + b_hh[col];
#pragma unroll
    for (int j = 0; j < 8; ++j) {
      gates[(size_t)(row0 + j + 8 * half) * G4 + col] = acc[0][i][j] + bias;
      gates[(size_t)(row0 + 16 + j + 8 * half) * G4 + col] = acc[1][i][j] + bias;
    }
  }
}

// pointwise LSTM cell
__global__ void k_lstm(const float* __restrict__ gates, const float* __restrict__ c_f32,
                       bf16* __restrict__ hn_bf, bf16* __restrict__ cn_bf) {
  int idx = blockIdx.x * blockDim.x + threadIdx.x;
  if (idx >= B_SZ * H_SZ) return;
  int b = idx >> 9, nn = idx & 511;
  const float* g = gates + (size_t)b * G4;
  float ig = g[nn], fg = g[H_SZ + nn], gg = g[2 * H_SZ + nn], og = g[3 * H_SZ + nn];
  float si = 1.f / (1.f + __expf(-ig));
  float sf = 1.f / (1.f + __expf(-fg));
  float so = 1.f / (1.f + __expf(-og));
  float cn = sf * c_f32[idx] + si * tanhf(gg);
  float hn = so * tanhf(cn);
  hn_bf[idx] = f2bf(hn);
  cn_bf[idx] = f2bf(cn);
}

// h2 = h_new @ hs_W^T ; c2 = c_new @ cs_W^T ; tanh epilogue + carries + outputs
__global__ __launch_bounds__(256) void k_proj(
    const bf16* __restrict__ hn_bf, const bf16* __restrict__ cn_bf,
    const bf16* __restrict__ Whs, const bf16* __restrict__ Wcs,
    const float* __restrict__ xi_f32, const float* __restrict__ X,
    float* __restrict__ c_f32, bf16* __restrict__ catA, bf16* __restrict__ xg,
    float* __restrict__ out, float* __restrict__ cs, int t) {
  const int lane = threadIdx.x & 31, w = threadIdx.x >> 5;
  const int row0 = blockIdx.x * 32;
  const int n_base = blockIdx.y * 256 + w * 32;  // 2 N-tiles per wave
  const int n = lane & 15, half = lane >> 4;
  floatx8 aH[2][2], aC[2][2];
#pragma unroll
  for (int mt = 0; mt < 2; ++mt)
#pragma unroll
    for (int i = 0; i < 2; ++i) { aH[mt][i] = (floatx8){}; aC[mt][i] = (floatx8){}; }

  for (int kt = 0; kt < H_SZ / 32; ++kt) {
    int k0 = kt * 32;
    bf16x16 ah0 = load_a_frag(hn_bf, H_SZ, row0, k0, lane);
    bf16x16 ah1 = load_a_frag(hn_bf, H_SZ, row0 + 16, k0, lane);
    bf16x16 ac0 = load_a_frag(cn_bf, H_SZ, row0, k0, lane);
    bf16x16 ac1 = load_a_frag(cn_bf, H_SZ, row0 + 16, k0, lane);
#pragma unroll
    for (int i = 0; i < 2; ++i) {
      bf16x16 bh = load_b_frag(Whs, H_SZ, n_base + i * 16, k0, lane);
      bf16x16 bc = load_b_frag(Wcs, H_SZ, n_base + i * 16, k0, lane);
      aH[0][i] = WMMA_BF16(ah0, bh, aH[0][i]);
      aH[1][i] = WMMA_BF16(ah1, bh, aH[1][i]);
      aC[0][i] = WMMA_BF16(ac0, bc, aC[0][i]);
      aC[1][i] = WMMA_BF16(ac1, bc, aC[1][i]);
    }
  }
#pragma unroll
  for (int mt = 0; mt < 2; ++mt) {
#pragma unroll
    for (int i = 0; i < 2; ++i) {
      int col = n_base + i * 16 + n;
#pragma unroll
      for (int j = 0; j < 8; ++j) {
        int row = row0 + mt * 16 + j + 8 * half;
        float h2 = aH[mt][i][j];
        float c2 = aC[mt][i][j];
        c_f32[(size_t)row * H_SZ + col] = c2;
        catA[(size_t)row * K1 + F_SZ + col] = f2bf(h2);
        catA[(size_t)row * K1 + 2 * F_SZ + col] = f2bf(c2);
        xg[(size_t)row * KG + F_SZ + col] = f2bf(h2);
        size_t oidx = ((size_t)row * T_SZ + t) * F_SZ + col;
        cs[oidx] = c2;
        float att = tanhf(h2 + c2);
        float o = tanhf(att + xi_f32[(size_t)row * F_SZ + col]);
        out[oidx] = X[oidx] + tanhf(o);
      }
    }
  }
}

// ---------------------------------------------------------------------------

extern "C" void kernel_launch(void* const* d_in, const int* in_sizes, int n_in,
                              void* d_out, int out_size, void* d_ws, size_t ws_size,
                              hipStream_t stream) {
  const float* X       = (const float*)d_in[0];
  const float* attn1_W = (const float*)d_in[1];
  const float* attn1_b = (const float*)d_in[2];
  const float* attn2_W = (const float*)d_in[3];
  const float* attn2_b = (const float*)d_in[4];
  const float* hs_W    = (const float*)d_in[5];
  const float* cs_W    = (const float*)d_in[6];
  const float* W_ih    = (const float*)d_in[7];
  const float* W_hh    = (const float*)d_in[8];
  const float* b_ih    = (const float*)d_in[9];
  const float* b_hh    = (const float*)d_in[10];

  float* out = (float*)d_out;
  float* cs  = out + (size_t)B_SZ * T_SZ * F_SZ;

  char* ws = (char*)d_ws;
  size_t off = 0;
  auto take = [&](size_t bytes) -> char* {
    char* p = ws + off;
    off += (bytes + 255) & ~(size_t)255;
    return p;
  };
  bf16*  Wb1    = (bf16*)take((size_t)N1P * K1 * 2);        // [96,1536] row-padded
  bf16*  Wb2    = (bf16*)take((size_t)F_SZ * N1P * 2);      // [512,96] col-padded
  bf16*  Wg     = (bf16*)take((size_t)G4 * KG * 2);         // [2048,1024]
  bf16*  Whs    = (bf16*)take((size_t)F_SZ * F_SZ * 2);
  bf16*  Wcs    = (bf16*)take((size_t)F_SZ * F_SZ * 2);
  bf16*  catA   = (bf16*)take((size_t)B_SZ * K1 * 2);       // [1024,1536] x|h|c
  bf16*  xg     = (bf16*)take((size_t)B_SZ * KG * 2);       // [1024,1024] xi|h
  float* xi_f32 = (float*)take((size_t)B_SZ * F_SZ * 4);
  float* gates  = (float*)take((size_t)B_SZ * G4 * 4);
  bf16*  hn_bf  = (bf16*)take((size_t)B_SZ * H_SZ * 2);
  bf16*  cn_bf  = (bf16*)take((size_t)B_SZ * H_SZ * 2);
  float* c_f32  = (float*)take((size_t)B_SZ * H_SZ * 4);

  k_cvt_padrow<<<(N1P * K1 + 255) / 256, 256, 0, stream>>>(attn1_W, Wb1, T_SZ + 1, N1P, K1);
  k_cvt_padcol<<<(F_SZ * N1P + 255) / 256, 256, 0, stream>>>(attn2_W, Wb2, F_SZ, T_SZ + 1, N1P);
  k_cvt_wg<<<(G4 * KG + 255) / 256, 256, 0, stream>>>(W_ih, W_hh, Wg);
  k_cvt_padcol<<<(F_SZ * F_SZ + 255) / 256, 256, 0, stream>>>(hs_W, Whs, F_SZ, F_SZ, F_SZ);
  k_cvt_padcol<<<(F_SZ * F_SZ + 255) / 256, 256, 0, stream>>>(cs_W, Wcs, F_SZ, F_SZ, F_SZ);
  k_init<<<(B_SZ * 1024 + 255) / 256, 256, 0, stream>>>(catA, xg, c_f32);

  for (int t = 0; t < T_SZ; ++t) {
    k_prep<<<(B_SZ * F_SZ + 255) / 256, 256, 0, stream>>>(X, catA, t);
    k_attn<<<B_SZ / 32, 256, 0, stream>>>(catA, Wb1, attn1_b, Wb2, attn2_b, X, xi_f32, xg, t);
    k_gates<<<dim3(B_SZ / 32, G4 / 512), 256, 0, stream>>>(xg, Wg, b_ih, b_hh, gates);
    k_lstm<<<(B_SZ * H_SZ + 255) / 256, 256, 0, stream>>>(gates, c_f32, hn_bf, cn_bf);
    k_proj<<<dim3(B_SZ / 32, F_SZ / 256), 256, 0, stream>>>(hn_bf, cn_bf, Whs, Wcs, xi_f32, X,
                                                            c_f32, catA, xg, out, cs, t);
  }
}
